// HydraMambaBlock_34265249088346
// MI455X (gfx1250) — compile-verified, matrix operand order checked by reading, not run on hardware
//
#include <hip/hip_runtime.h>
#include <hip/hip_bf16.h>

// ---------------- problem constants (match reference) ----------------
constexpr int SEQ_L   = 2048;
constexpr int DMODEL  = 256;
constexpr int NSTATE  = 16;
constexpr int DINNER  = 512;   // EXPAND * DMODEL
constexpr int HEADDIM = 64;    // DINNER / NHEADS
constexpr int DIN     = 2 * DINNER + 2 * NSTATE + 8; // 1064
constexpr int HIDDIM  = 1024;  // 4 * DMODEL
constexpr int ROWS    = 4 * SEQ_L; // B*L = 8192

// ---------------- WMMA vector types ----------------
typedef __bf16 bf16_t;
typedef __attribute__((ext_vector_type(16))) __bf16 v16bf;
typedef __attribute__((ext_vector_type(8)))  float  v8f;

static __device__ __forceinline__ bf16_t tobf(float f) {
  unsigned u = __builtin_bit_cast(unsigned, f);
  unsigned r = u + 0x7FFFu + ((u >> 16) & 1u);   // round-to-nearest-even
  unsigned short h = (unsigned short)(r >> 16);
  return __builtin_bit_cast(bf16_t, h);
}

static __device__ __forceinline__ int mapRowFlip(int r) {
  int b = r / SEQ_L, l = r % SEQ_L;
  return b * SEQ_L + (SEQ_L - 1 - l);
}

// ---- CDNA5 async memory->LDS (ASYNCcnt-tracked, no VGPR round trip) ----
static __device__ __forceinline__ void wait_async0() {
#if __has_builtin(__builtin_amdgcn_s_wait_asynccnt)
  __builtin_amdgcn_s_wait_asynccnt(0);
#else
  asm volatile("s_wait_asynccnt 0x0" ::: "memory");
#endif
}

// the immediate offset is added to both the LDS and the 64-bit global address
#define ASYNC_LDS_B128(ldsoff, gaddr, OFFS)                                   \
  asm volatile("global_load_async_to_lds_b128 %0, %1, off offset:" #OFFS      \
               :: "v"(ldsoff), "v"(gaddr) : "memory")

// ---------------- weight convert: W[K,N] fp32 -> Wt[N,K] bf16 ----------------
__global__ void convert_wt_kernel(const float* __restrict__ W, bf16_t* __restrict__ Wt,
                                  int Kd, int N)
{
  size_t idx = (size_t)blockIdx.x * blockDim.x + threadIdx.x;
  if (idx >= (size_t)Kd * N) return;
  int n = (int)(idx / Kd), k = (int)(idx % Kd);
  Wt[idx] = tobf(W[(size_t)k * N + n]);
}

// ---------------- tiled bf16 WMMA GEMM: C = A[MxK] * W[KxN] (+bias)(+resid) -------------
// block tile: 32(M) x 256(N) x 32(K); 8 waves; each wave: 16M x 64N (4 accumulators)
// double-buffered LDS; B tile staged via async-to-LDS DMA overlapping the WMMAs.
#define GF_FLIPA   1
#define GF_FLIPOUT 2
#define GF_ACCUM   4

constexpr int BLK_M = 32, BLK_N = 256, BLK_K = 32;
constexpr int LSTR  = 40;   // LDS element stride (bf16) -> 80B, 16B aligned, bank-spread

__global__ __launch_bounds__(256) void wmma_gemm_kernel(
    const float* __restrict__ A, const bf16_t* __restrict__ Wt,
    float* __restrict__ Cmat, const float* __restrict__ bias,
    const float* __restrict__ resid, int M, int N, int Kd, int flags)
{
  __shared__ bf16_t As[2][BLK_M * LSTR];    // ~5 KB
  __shared__ bf16_t Bs[2][BLK_N * LSTR];    // ~40 KB

  const int tid  = threadIdx.x;
  const int lane = tid & 31;
  const int w    = tid >> 5;             // wave 0..7
  const int hlf  = lane >> 4;            // K-half select (A/B frags); C row +8
  const int i16  = lane & 15;
  const int waveM = w & 1;               // 2 waves along M
  const int waveN = w >> 1;              // 4 waves along N

  const int rowBase = blockIdx.y * BLK_M;
  const int nBase   = blockIdx.x * BLK_N;

  // --- staging assignments ---
  const int arow = tid >> 3;             // 0..31 (8 threads per A row)
  const int acol = (tid & 7) * 4;        // 4 floats each
  int aPhysRow = rowBase + arow;
  if (flags & GF_FLIPA) aPhysRow = mapRowFlip(aPhysRow);
  const float* Aload = A + (size_t)aPhysRow * Kd + acol;

  const int  bn   = tid;                 // one B row (of Wt) per thread
  const bool bnOK = (nBase + bn) < N;
  const bf16_t* Bload = Wt + (size_t)(bnOK ? (nBase + bn) : 0) * Kd;

  // LDS byte offsets for the async engine (generic ptr low 32 bits == LDS offset)
  const unsigned bdst0 = (unsigned)(uintptr_t)&Bs[0][bn * LSTR];
  const unsigned bdst1 = (unsigned)(uintptr_t)&Bs[1][bn * LSTR];

  // stage tile (A: fp32->bf16 via VALU + ds_store; B: async DMA, 4x16B per thread)
  auto stage = [&](int k0, int buf) {
    const float4 av = *(const float4*)(Aload + k0);
    bf16_t* ad = &As[buf][arow * LSTR + acol];
    ad[0] = tobf(av.x); ad[1] = tobf(av.y); ad[2] = tobf(av.z); ad[3] = tobf(av.w);
    if (bnOK) {
      unsigned  bdst = buf ? bdst1 : bdst0;
      unsigned long long ga = (unsigned long long)(Bload + k0);
      ASYNC_LDS_B128(bdst, ga, 0);
      ASYNC_LDS_B128(bdst, ga, 16);
      ASYNC_LDS_B128(bdst, ga, 32);
      ASYNC_LDS_B128(bdst, ga, 48);
    } else {
      uint4 z = {0u, 0u, 0u, 0u};
#pragma unroll
      for (int q = 0; q < 4; ++q) *(uint4*)&Bs[buf][bn * LSTR + q * 8] = z;
    }
  };

  v8f acc[4];
#pragma unroll
  for (int j = 0; j < 4; ++j) acc[j] = (v8f){0.f,0.f,0.f,0.f,0.f,0.f,0.f,0.f};

  stage(0, 0);
  int cur = 0;

  for (int k0 = 0; k0 < Kd; k0 += BLK_K) {
    wait_async0();            // our async stores for `cur` complete
    __syncthreads();          // everyone's `cur` tile staged

    const bool more = (k0 + BLK_K) < Kd;
    if (more) {
      __builtin_prefetch(Aload + k0 + BLK_K, 0, 1);      // global_prefetch_b8
      stage(k0 + BLK_K, cur ^ 1);                        // DMA overlaps WMMAs below
    }

    // ---- fragments + 4 WMMAs from `cur` ----
    const int kb = hlf * 8;
    v16bf af;
    {
      const bf16_t* ap = &As[cur][(waveM * 16 + i16) * LSTR];
#pragma unroll
      for (int e = 0; e < 8; ++e) { af[e] = ap[kb + e]; af[e + 8] = ap[kb + 16 + e]; }
    }
#pragma unroll
    for (int j = 0; j < 4; ++j) {
      const bf16_t* bp = &Bs[cur][(waveN * 64 + j * 16 + i16) * LSTR];
      v16bf bfr;
#pragma unroll
      for (int e = 0; e < 8; ++e) { bfr[e] = bp[kb + e]; bfr[e + 8] = bp[kb + 16 + e]; }
      acc[j] = __builtin_amdgcn_wmma_f32_16x16x32_bf16(
          false, af, false, bfr, (short)0, acc[j], false, false);
    }
    __syncthreads();          // all reads of `cur` done before it is restaged
    cur ^= 1;
  }

  // ---- epilogue ----
#pragma unroll
  for (int j = 0; j < 4; ++j) {
    int col = nBase + waveN * 64 + j * 16 + i16;
    if (col >= N) continue;
#pragma unroll
    for (int r = 0; r < 8; ++r) {
      int row = rowBase + waveM * 16 + r + 8 * hlf;
      float v = acc[j][r];
      if (bias) v += bias[col];
      int orow = (flags & GF_FLIPOUT) ? mapRowFlip(row) : row;
      size_t oidx = (size_t)orow * N + col;
      if (resid) v += resid[oidx];
      if (flags & GF_ACCUM) Cmat[oidx] += v;
      else                  Cmat[oidx]  = v;
    }
  }
}

// ---------------- LayerNorm over last dim (DMODEL=256), one block per row ----------------
__global__ __launch_bounds__(256) void ln_kernel(
    const float* __restrict__ in, const float* __restrict__ g,
    const float* __restrict__ b, float* __restrict__ out)
{
  __shared__ float red[256];
  const int row = blockIdx.x, tid = threadIdx.x;
  float v = in[(size_t)row * DMODEL + tid];
  red[tid] = v; __syncthreads();
  for (int s = 128; s > 0; s >>= 1) { if (tid < s) red[tid] += red[tid + s]; __syncthreads(); }
  float mean = red[0] * (1.f / DMODEL);
  __syncthreads();
  float d = v - mean;
  red[tid] = d * d; __syncthreads();
  for (int s = 128; s > 0; s >>= 1) { if (tid < s) red[tid] += red[tid + s]; __syncthreads(); }
  float var = red[0] * (1.f / DMODEL);
  out[(size_t)row * DMODEL + tid] = d * rsqrtf(var + 1e-5f) * g[tid] + b[tid];
}

// ---------------- Mamba2 selective scan: one block per (b,h); 1024 threads = 64P x 16N ----
__global__ __launch_bounds__(1024) void scan_kernel(
    const float* __restrict__ proj, const float* __restrict__ dt_bias,
    const float* __restrict__ a_log, const float* __restrict__ d_skip,
    float* __restrict__ y)
{
  const int b = blockIdx.x >> 3;
  const int h = blockIdx.x & 7;
  const int p = threadIdx.x >> 4;         // 0..63
  const int n = threadIdx.x & 15;         // 0..15

  const float a   = -__expf(a_log[h]);
  const float db  = dt_bias[h];
  const float dsk = d_skip[h];
  float st = 0.f;

  const float* base = proj + (size_t)b * SEQ_L * DIN;
  for (int t = 0; t < SEQ_L; ++t) {
    const float* rp = base + (size_t)t * DIN;
    float dtraw = rp[2 * DINNER + 2 * NSTATE + h] + db;
    float dt = (dtraw > 20.f) ? dtraw : log1pf(__expf(dtraw));   // softplus
    float da = __expf(dt * a);
    float xp = rp[DINNER + h * HEADDIM + p];
    float bn = rp[2 * DINNER + n];
    float cn = rp[2 * DINNER + NSTATE + n];
    st = da * st + (dt * xp) * bn;
    float c = st * cn;
    c += __shfl_xor(c, 1, 16);
    c += __shfl_xor(c, 2, 16);
    c += __shfl_xor(c, 4, 16);
    c += __shfl_xor(c, 8, 16);
    if (n == 0) {
      float z  = rp[h * HEADDIM + p];
      float yv = (c + dsk * xp) * (z / (1.f + __expf(-z)));      // +D*x, *silu(z)
      y[((size_t)(b * SEQ_L + t)) * DINNER + h * HEADDIM + p] = yv;
    }
  }
}

// ---------------- fused: tok += gelu(dwconv(x_ln)) + points (x1 in place) ----------------
__global__ void convgelu_kernel(
    const float* __restrict__ xln, const float* __restrict__ cw,
    const float* __restrict__ cb,  const float* __restrict__ points,
    float* __restrict__ tok)
{
  size_t idx = (size_t)blockIdx.x * blockDim.x + threadIdx.x;
  if (idx >= (size_t)ROWS * DMODEL) return;
  int c = (int)(idx & (DMODEL - 1));
  size_t row = idx >> 8;
  int bb = (int)(row / SEQ_L), l = (int)(row % SEQ_L);
  float s = cb[c];
#pragma unroll
  for (int k = 0; k < 3; ++k) {
    int ls = l + k - 1;
    if (ls >= 0 && ls < SEQ_L)
      s += xln[((size_t)(bb * SEQ_L + ls)) * DMODEL + c] * cw[c * 3 + k];
  }
  float gel = 0.5f * s * (1.f + erff(s * 0.70710678118654752f));
  tok[idx] = tok[idx] + gel + points[idx];
}

// ---------------- BatchNorm (train-mode) stats + apply ----------------
__global__ __launch_bounds__(256) void bnstats_kernel(
    const float* __restrict__ hbuf, float* __restrict__ stats)
{
  __shared__ float s1[256], s2[256];
  const int ch = blockIdx.x, tid = threadIdx.x;
  float sum = 0.f, sq = 0.f;
  for (int r = tid; r < ROWS; r += 256) {
    float v = hbuf[(size_t)r * HIDDIM + ch];
    sum += v; sq += v * v;
  }
  s1[tid] = sum; s2[tid] = sq; __syncthreads();
  for (int s = 128; s > 0; s >>= 1) {
    if (tid < s) { s1[tid] += s1[tid + s]; s2[tid] += s2[tid + s]; }
    __syncthreads();
  }
  if (tid == 0) {
    float m = s1[0] * (1.f / ROWS);
    stats[ch]          = m;
    stats[HIDDIM + ch] = s2[0] * (1.f / ROWS) - m * m;
  }
}

__global__ void bnapply_kernel(
    float* __restrict__ hbuf, const float* __restrict__ stats,
    const float* __restrict__ g, const float* __restrict__ b)
{
  size_t idx = (size_t)blockIdx.x * blockDim.x + threadIdx.x;
  if (idx >= (size_t)ROWS * HIDDIM) return;
  int ch = (int)(idx & (HIDDIM - 1));
  float m = stats[ch], v = stats[HIDDIM + ch];
  float x = (hbuf[idx] - m) * rsqrtf(v + 1e-5f) * g[ch] + b[ch];
  hbuf[idx] = x > 0.f ? x : 0.f;
}

// ---------------- launcher ----------------
extern "C" void kernel_launch(void* const* d_in, const int* in_sizes, int n_in,
                              void* d_out, int out_size, void* d_ws, size_t ws_size,
                              hipStream_t stream) {
  const float* points      = (const float*)d_in[0];
  const float* fssm_in_w   = (const float*)d_in[1];
  const float* fssm_dt_b   = (const float*)d_in[2];
  const float* fssm_a_log  = (const float*)d_in[3];
  const float* fssm_d      = (const float*)d_in[4];
  const float* fssm_out_w  = (const float*)d_in[5];
  const float* bssm_in_w   = (const float*)d_in[6];
  const float* bssm_dt_b   = (const float*)d_in[7];
  const float* bssm_a_log  = (const float*)d_in[8];
  const float* bssm_d      = (const float*)d_in[9];
  const float* bssm_out_w  = (const float*)d_in[10];
  const float* ln1_g       = (const float*)d_in[11];
  const float* ln1_b       = (const float*)d_in[12];
  const float* conv_w      = (const float*)d_in[13];
  const float* conv_b      = (const float*)d_in[14];
  const float* mlp_w1      = (const float*)d_in[15];
  const float* mlp_b1      = (const float*)d_in[16];
  const float* bn_g        = (const float*)d_in[17];
  const float* bn_b        = (const float*)d_in[18];
  const float* mlp_w2      = (const float*)d_in[19];
  const float* mlp_b2      = (const float*)d_in[20];
  const float* ln2_g       = (const float*)d_in[21];
  const float* ln2_b       = (const float*)d_in[22];
  float* out = (float*)d_out;

  float* ws = (float*)d_ws;
  size_t off = 0;
  float* x_ln  = ws + off; off += (size_t)ROWS * DMODEL;
  float* proj  = ws + off; off += (size_t)ROWS * DIN;
  float* yssm  = ws + off; off += (size_t)ROWS * DINNER;
  float* tok   = ws + off; off += (size_t)ROWS * DMODEL;
  float* ln2bf = ws + off; off += (size_t)ROWS * DMODEL;
  float* hbuf  = ws + off; off += (size_t)ROWS * HIDDIM;
  float* stats = ws + off; off += 2 * HIDDIM;

  // bf16 transposed weight cache
  bf16_t* wtb = (bf16_t*)(ws + off);
  size_t boff = 0;
  bf16_t* wt_fin  = wtb + boff; boff += (size_t)DMODEL * DIN;
  bf16_t* wt_bin  = wtb + boff; boff += (size_t)DMODEL * DIN;
  bf16_t* wt_fout = wtb + boff; boff += (size_t)DINNER * DMODEL;
  bf16_t* wt_bout = wtb + boff; boff += (size_t)DINNER * DMODEL;
  bf16_t* wt_w1   = wtb + boff; boff += (size_t)DMODEL * HIDDIM;
  bf16_t* wt_w2   = wtb + boff; boff += (size_t)HIDDIM * DMODEL;

  auto cvt = [&](const float* W, bf16_t* Wt, int Kd, int N) {
    size_t tot = (size_t)Kd * N;
    convert_wt_kernel<<<(unsigned)((tot + 255) / 256), 256, 0, stream>>>(W, Wt, Kd, N);
  };
  cvt(fssm_in_w,  wt_fin,  DMODEL, DIN);
  cvt(bssm_in_w,  wt_bin,  DMODEL, DIN);
  cvt(fssm_out_w, wt_fout, DINNER, DMODEL);
  cvt(bssm_out_w, wt_bout, DINNER, DMODEL);
  cvt(mlp_w1,     wt_w1,   DMODEL, HIDDIM);
  cvt(mlp_w2,     wt_w2,   HIDDIM, DMODEL);

  const dim3 gIn((DIN    + BLK_N - 1) / BLK_N, ROWS / BLK_M);  // 5 x 256
  const dim3 gD ((DMODEL + BLK_N - 1) / BLK_N, ROWS / BLK_M);  // 1 x 256
  const dim3 gH ((HIDDIM + BLK_N - 1) / BLK_N, ROWS / BLK_M);  // 4 x 256

  // x = LN1(points)
  ln_kernel<<<ROWS, 256, 0, stream>>>(points, ln1_g, ln1_b, x_ln);

  // forward SSM
  wmma_gemm_kernel<<<gIn, 256, 0, stream>>>(x_ln, wt_fin, proj, nullptr, nullptr,
                                            ROWS, DIN, DMODEL, 0);
  scan_kernel<<<32, 1024, 0, stream>>>(proj, fssm_dt_b, fssm_a_log, fssm_d, yssm);
  wmma_gemm_kernel<<<gD, 256, 0, stream>>>(yssm, wt_fout, tok, nullptr, nullptr,
                                           ROWS, DMODEL, DINNER, 0);

  // backward SSM (flip handled by index mapping)
  wmma_gemm_kernel<<<gIn, 256, 0, stream>>>(x_ln, wt_bin, proj, nullptr, nullptr,
                                            ROWS, DIN, DMODEL, GF_FLIPA);
  scan_kernel<<<32, 1024, 0, stream>>>(proj, bssm_dt_b, bssm_a_log, bssm_d, yssm);
  wmma_gemm_kernel<<<gD, 256, 0, stream>>>(yssm, wt_bout, tok, nullptr, nullptr,
                                           ROWS, DMODEL, DINNER, GF_FLIPOUT | GF_ACCUM);

  // tok += gelu(dwconv(x)) ; x1 = tok + points
  {
    size_t tot = (size_t)ROWS * DMODEL;
    convgelu_kernel<<<(unsigned)((tot + 255) / 256), 256, 0, stream>>>(
        x_ln, conv_w, conv_b, points, tok);
  }

  // MLP branch
  ln_kernel<<<ROWS, 256, 0, stream>>>(tok, ln2_g, ln2_b, ln2bf);
  wmma_gemm_kernel<<<gH, 256, 0, stream>>>(ln2bf, wt_w1, hbuf, mlp_b1, nullptr,
                                           ROWS, HIDDIM, DMODEL, 0);
  bnstats_kernel<<<HIDDIM, 256, 0, stream>>>(hbuf, stats);
  {
    size_t tot2 = (size_t)ROWS * HIDDIM;
    bnapply_kernel<<<(unsigned)((tot2 + 255) / 256), 256, 0, stream>>>(
        hbuf, stats, bn_g, bn_b);
  }
  // out = relu_bn_h @ w2 + b2 + x1
  wmma_gemm_kernel<<<gD, 256, 0, stream>>>(hbuf, wt_w2, out, mlp_b2, tok,
                                           ROWS, DMODEL, HIDDIM, 0);
}
